// SpatialLinearAttention_32169305047464
// MI455X (gfx1250) — compile-verified, matrix operand order checked by reading, not run on hardware
//
#include <hip/hip_runtime.h>
#include <hip/hip_bf16.h>

typedef __attribute__((ext_vector_type(16))) __bf16 v16bf;
typedef __attribute__((ext_vector_type(8)))  __bf16 v8bf;
typedef __attribute__((ext_vector_type(8)))  float  v8f;
typedef unsigned long long u64;

// ---------- constants ----------
#define BF    32        // b*f
#define NSP   4096      // h*w
#define C     256
#define HEADS 8
#define DH    32        // head dim
#define MROWS (BF*NSP)  // 131072 GEMM rows
#define KS    264       // padded LDS row stride (elements) for 256-wide tiles
#define TS    68        // padded LDS row stride for transpose buffer

__device__ inline v16bf cat8(v8bf lo, v8bf hi) {
  return __builtin_shufflevector(lo, hi, 0,1,2,3,4,5,6,7,8,9,10,11,12,13,14,15);
}
// A/B fragment: 8 contiguous bf16 at p (K+0..7) and 8 at p+16 (K+16..23);
// caller applies the lane>=16 "+8" K offset. Two 16B loads.
__device__ inline v16bf ldfrag(const __bf16* p) {
  return cat8(*(const v8bf*)p, *(const v8bf*)(p + 16));
}
__device__ inline v8f wmma_bf16(v16bf a, v16bf b, v8f c) {
  return __builtin_amdgcn_wmma_f32_16x16x32_bf16(false, a, false, b, (short)0, c, false, false);
}

// ---------- K0: weight transpose/convert + zero Z ----------
// wqkvT[mat][ncol][k] bf16 (mat 0..2), woT[ncol][k] bf16, zsum[BF*HEADS*DH]=0
__global__ __launch_bounds__(256) void k0_prep(const float* Wq, const float* Wk,
                                               const float* Wv, const float* Wo,
                                               __bf16* wqkvT, __bf16* woT, float* zsum) {
  int i = blockIdx.x * 256 + threadIdx.x;            // 0 .. 4*65536-1
  if (i < BF * HEADS * DH) zsum[i] = 0.f;
  int mat = i >> 16;
  int r   = (i >> 8) & 255;                          // ncol
  int k   = i & 255;
  const float* src = (mat == 0) ? Wq : (mat == 1) ? Wk : (mat == 2) ? Wv : Wo;
  float v = src[k * 256 + r];
  if (mat < 3) wqkvT[i] = (__bf16)v;
  else         woT[i & 65535] = (__bf16)v;
}

// ---------- K1: qkv projections + softmaxes + head transposes ----------
// grid = MROWS/64 blocks, 256 threads (8 waves). Wave = 16 rows x 32 cols (one head).
__global__ __launch_bounds__(256) void k1_proj(const float* x, const __bf16* wqkvT,
                                               __bf16* qs, __bf16* ekT, __bf16* vT,
                                               float* zsum) {
  __shared__ __bf16 xs[64 * KS];      // x tile, bf16
  __shared__ __bf16 tbuf[64 * TS];    // transpose buffer [d(2 heads)][n]
  __shared__ float  zbuf[64];

  const int tid  = threadIdx.x;
  const int lane = tid & 31;
  const int wave = tid >> 5;
  const int rowbase = blockIdx.x * 64;       // global GEMM row
  const int bfi   = rowbase >> 12;
  const int nloc0 = rowbase & (NSP - 1);
  const int am   = lane & 15;
  const int aoff = (lane & 16) ? 8 : 0;
  const int rofs = (lane >> 4) << 3;
  const int mrow = (wave & 3) * 16;          // wave's row quarter in tile
  const int hsel = wave >> 2;                // which head of the pair

  // stage X tile: 64x256 f32 -> bf16  (16 float4 per thread)
  {
    const float4* xp = reinterpret_cast<const float4*>(x + (size_t)rowbase * C);
#pragma unroll
    for (int p = 0; p < 16; ++p) {
      float4 f = xp[p * 256 + tid];
      int fidx = (p * 256 + tid) * 4;
      int r = fidx >> 8, c = fidx & 255;
      __bf16* d = &xs[r * KS + c];
      d[0] = (__bf16)f.x; d[1] = (__bf16)f.y; d[2] = (__bf16)f.z; d[3] = (__bf16)f.w;
    }
  }
  __syncthreads();

  for (int mat = 0; mat < 3; ++mat) {
    for (int hb = 0; hb < HEADS; hb += 2) {
      const int head = hb + hsel;
      v8f acc0 = {}, acc1 = {};
      const __bf16* wb = wqkvT + ((size_t)(mat * 256 + (head << 5))) * 256;
#pragma unroll
      for (int kb = 0; kb < 256; kb += 32) {
        v16bf A  = ldfrag(&xs[(mrow + am) * KS + kb + aoff]);
        const __bf16* b0 = wb + (am << 8) + kb + aoff;
        v16bf B0 = ldfrag(b0);
        v16bf B1 = ldfrag(b0 + (16 << 8));
        acc0 = wmma_bf16(A, B0, acc0);
        acc1 = wmma_bf16(A, B1, acc1);
      }

      if (mat == 0) {
        // q: softmax over the 32 head channels (row-local)
        const size_t rowg = (size_t)(rowbase + mrow + rofs);
        __bf16* q0 = qs + rowg * C + (head << 5) + am;
#pragma unroll
        for (int r = 0; r < 8; ++r) {
          float e0 = __expf(acc0[r]);
          float e1 = __expf(acc1[r]);
          float s = e0 + e1;
          s += __shfl_xor(s, 1, 16); s += __shfl_xor(s, 2, 16);
          s += __shfl_xor(s, 4, 16); s += __shfl_xor(s, 8, 16);
          float inv = 1.0f / s;
          q0[(size_t)r * C]      = (__bf16)(e0 * inv);
          q0[(size_t)r * C + 16] = (__bf16)(e1 * inv);
        }
      } else if (mat == 1) {
        // k: exp, accumulate column (n-) sums into Z, write head-transposed exp(k)
        float e0[8], e1[8], cs0 = 0.f, cs1 = 0.f;
#pragma unroll
        for (int r = 0; r < 8; ++r) {
          e0[r] = __expf(acc0[r]); e1[r] = __expf(acc1[r]);
          cs0 += e0[r]; cs1 += e1[r];
        }
        cs0 += __shfl_xor(cs0, 16, 32);
        cs1 += __shfl_xor(cs1, 16, 32);
        __syncthreads();
        if (tid < 64) zbuf[tid] = 0.f;
        __syncthreads();
        if (lane < 16) {
          atomicAdd(&zbuf[(hsel << 5) + am],      cs0);
          atomicAdd(&zbuf[(hsel << 5) + 16 + am], cs1);
        }
        const int dl0 = (hsel << 5) + am, dl1 = dl0 + 16, nl = mrow + rofs;
#pragma unroll
        for (int r = 0; r < 8; ++r) {
          tbuf[dl0 * TS + nl + r] = (__bf16)e0[r];
          tbuf[dl1 * TS + nl + r] = (__bf16)e1[r];
        }
        __syncthreads();
        if (tid < 64)
          atomicAdd(&zsum[((bfi << 3) + hb + (tid >> 5)) * DH + (tid & 31)], zbuf[tid]);
        {
          const int d = tid >> 2, ng = (tid & 3) << 4;
          const u64* sp = (const u64*)&tbuf[d * TS + ng];
          u64 a0 = sp[0], a1 = sp[1], a2 = sp[2], a3 = sp[3];
          __bf16* dst = ekT + ((size_t)(((bfi << 3) + hb + (d >> 5)) * DH + (d & 31))) * NSP
                          + nloc0 + ng;
          u64* dp = (u64*)dst;
          dp[0] = a0; dp[1] = a1; dp[2] = a2; dp[3] = a3;
        }
        __syncthreads();
      } else {
        // v: head-transposed bf16
        __syncthreads();
        const int dl0 = (hsel << 5) + am, dl1 = dl0 + 16, nl = mrow + rofs;
#pragma unroll
        for (int r = 0; r < 8; ++r) {
          tbuf[dl0 * TS + nl + r] = (__bf16)acc0[r];
          tbuf[dl1 * TS + nl + r] = (__bf16)acc1[r];
        }
        __syncthreads();
        {
          const int d = tid >> 2, ng = (tid & 3) << 4;
          const u64* sp = (const u64*)&tbuf[d * TS + ng];
          u64 a0 = sp[0], a1 = sp[1], a2 = sp[2], a3 = sp[3];
          __bf16* dst = vT + ((size_t)(((bfi << 3) + hb + (d >> 5)) * DH + (d & 31))) * NSP
                         + nloc0 + ng;
          u64* dp = (u64*)dst;
          dp[0] = a0; dp[1] = a1; dp[2] = a2; dp[3] = a3;
        }
        __syncthreads();
      }
    }
  }
}

// ---------- K2: context per (bf,head): ct[d][e] = sum_n ek[d,n] v[e,n], /Z[d] ----------
// grid = BF*HEADS blocks; 8 waves split K=4096; operands row-major [32][4096] ->
// fragments load straight from global with b128s. Output ctT[e][d] bf16.
__global__ __launch_bounds__(256) void k2_ctx(const __bf16* ekT, const __bf16* vT,
                                              const float* zsum, __bf16* ct) {
  __shared__ float cbuf[DH * DH];
  const int tid = threadIdx.x, lane = tid & 31, wave = tid >> 5;
  const int bh = blockIdx.x;
  const __bf16* ek = ekT + (size_t)bh * DH * NSP;
  const __bf16* vv = vT  + (size_t)bh * DH * NSP;
#pragma unroll
  for (int i = 0; i < 4; ++i) cbuf[tid * 4 + i] = 0.f;
  __syncthreads();

  const int am = lane & 15, aoff = (lane & 16) ? 8 : 0, rofs = (lane >> 4) << 3;
  v8f a00 = {}, a01 = {}, a10 = {}, a11 = {};
  const int kstart = wave * 512;
  for (int kb = kstart; kb < kstart + 512; kb += 32) {
    const __bf16* pa0 = ek + (size_t)am * NSP + kb + aoff;
    v16bf A0 = ldfrag(pa0);
    v16bf A1 = ldfrag(pa0 + 16 * NSP);
    const __bf16* pb0 = vv + (size_t)am * NSP + kb + aoff;
    v16bf B0 = ldfrag(pb0);
    v16bf B1 = ldfrag(pb0 + 16 * NSP);
    a00 = wmma_bf16(A0, B0, a00);
    a01 = wmma_bf16(A0, B1, a01);
    a10 = wmma_bf16(A1, B0, a10);
    a11 = wmma_bf16(A1, B1, a11);
  }
#pragma unroll
  for (int r = 0; r < 8; ++r) {
    atomicAdd(&cbuf[(r + rofs) * DH + am],            a00[r]);
    atomicAdd(&cbuf[(r + rofs) * DH + 16 + am],       a01[r]);
    atomicAdd(&cbuf[(16 + r + rofs) * DH + am],       a10[r]);
    atomicAdd(&cbuf[(16 + r + rofs) * DH + 16 + am],  a11[r]);
  }
  __syncthreads();
  const float* Z = zsum + bh * DH;
  __bf16* out = ct + (size_t)bh * DH * DH;   // [e][d]
#pragma unroll
  for (int i = 0; i < 4; ++i) {
    int j = tid * 4 + i;
    int e = j >> 5, d = j & 31;
    out[j] = (__bf16)(cbuf[d * DH + e] / Z[d]);
  }
}

// ---------- K3: fused  attn = q @ ctT  (per head, into LDS)  then  out = attn @ WoT ----------
// grid = MROWS/64 blocks; 256 threads.
__global__ __launch_bounds__(256) void k3_out(const __bf16* qs, const __bf16* ct,
                                              const __bf16* woT, float* out) {
  __shared__ __bf16 as[64 * KS];
  const int tid = threadIdx.x, lane = tid & 31, wave = tid >> 5;
  const int rowbase = blockIdx.x * 64;
  const int bfi = rowbase >> 12;
  const int am = lane & 15, aoff = (lane & 16) ? 8 : 0, rofs = (lane >> 4) << 3;

  // phase A: per-head 16x32 attn tiles (K = 32, one chunk)
  {
    const int mrow = (wave & 3) * 16;
    const int hs = (wave >> 2) * 4;
#pragma unroll
    for (int h = 0; h < 4; ++h) {
      const int head = hs + h;
      const __bf16* qp = qs + ((size_t)(rowbase + mrow + am)) * C + (head << 5) + aoff;
      v16bf A = ldfrag(qp);
      const __bf16* cp = ct + ((size_t)((bfi << 3) + head) << 10) + (am << 5) + aoff;
      v16bf B0 = ldfrag(cp);
      v16bf B1 = ldfrag(cp + (16 << 5));
      v8f c0 = {}, c1 = {};
      c0 = wmma_bf16(A, B0, c0);
      c1 = wmma_bf16(A, B1, c1);
      __bf16* d0 = &as[(mrow + rofs) * KS + (head << 5) + am];
#pragma unroll
      for (int r = 0; r < 8; ++r) {
        d0[r * KS]      = (__bf16)c0[r];
        d0[r * KS + 16] = (__bf16)c1[r];
      }
    }
  }
  __syncthreads();

  // phase B: 64x256 attn tile (LDS) @ WoT -> f32 out, wave owns 32 output cols
  const int cb = wave << 5;
  for (int mblk = 0; mblk < 4; ++mblk) {
    v8f c0 = {}, c1 = {};
#pragma unroll
    for (int kb = 0; kb < 256; kb += 32) {
      v16bf A = ldfrag(&as[(mblk * 16 + am) * KS + kb + aoff]);
      const __bf16* b0 = woT + ((cb + am) << 8) + kb + aoff;
      v16bf B0 = ldfrag(b0);
      v16bf B1 = ldfrag(b0 + (16 << 8));
      c0 = wmma_bf16(A, B0, c0);
      c1 = wmma_bf16(A, B1, c1);
    }
    float* o0 = out + ((size_t)(rowbase + mblk * 16 + rofs)) * C + cb + am;
#pragma unroll
    for (int r = 0; r < 8; ++r) {
      o0[(size_t)r * C]      = c0[r];
      o0[(size_t)r * C + 16] = c1[r];
    }
  }
}

// ---------- host ----------
extern "C" void kernel_launch(void* const* d_in, const int* in_sizes, int n_in,
                              void* d_out, int out_size, void* d_ws, size_t ws_size,
                              hipStream_t stream) {
  const float* x  = (const float*)d_in[0];
  const float* Wq = (const float*)d_in[1];
  const float* Wk = (const float*)d_in[2];
  const float* Wv = (const float*)d_in[3];
  const float* Wo = (const float*)d_in[4];
  float* outp = (float*)d_out;

  unsigned char* w = (unsigned char*)d_ws;
  __bf16* wqkvT = (__bf16*)w;                w += (size_t)3 * 65536 * 2;
  __bf16* woT   = (__bf16*)w;                w += (size_t)65536 * 2;
  float*  zsum  = (float*)w;                 w += (size_t)BF * HEADS * DH * 4;
  __bf16* ctb   = (__bf16*)w;                w += (size_t)BF * HEADS * DH * DH * 2;
  __bf16* qsb   = (__bf16*)w;                w += (size_t)MROWS * C * 2;
  __bf16* ekT   = (__bf16*)w;                w += (size_t)BF * HEADS * DH * NSP * 2;
  __bf16* vT    = (__bf16*)w;

  k0_prep<<<1024, 256, 0, stream>>>(Wq, Wk, Wv, Wo, wqkvT, woT, zsum);
  k1_proj<<<MROWS / 64, 256, 0, stream>>>(x, wqkvT, qsb, ekT, vT, zsum);
  k2_ctx<<<BF * HEADS, 256, 0, stream>>>(ekT, vT, zsum, ctb);
  k3_out<<<MROWS / 64, 256, 0, stream>>>(qsb, ctb, woT, outp);
}
